// AttentionPairBias_10428180595078
// MI455X (gfx1250) — matrix operand order repaired and checked by reading, not compiled, not run on hardware
//
#include <hip/hip_runtime.h>
#include <hip/hip_bf16.h>

// ---------------------------------------------------------------------------
// AttentionPairBias for MI455X (gfx1250, wave32, WMMA 16x16x32 bf16)
// B=1, N=1024, c_a=768, c_z=128, H=16, c=48
// ---------------------------------------------------------------------------

typedef __attribute__((ext_vector_type(16))) __bf16 bf16x16;
typedef __attribute__((ext_vector_type(8)))  float  f32x8;

__device__ __forceinline__ f32x8 wmma_bf16(bf16x16 a, bf16x16 b, f32x8 c) {
    // emits v_wmma_f32_16x16x32_bf16
    return __builtin_amdgcn_wmma_f32_16x16x32_bf16(
        /*neg_a=*/false, a, /*neg_b=*/false, b,
        /*c_mod=*/(short)0, c, /*reuse_a=*/false, /*reuse_b=*/false);
}

#define N_TOK   1024
#define C_A     768
#define C_Z     128
#define NHEAD   16
#define C_HEAD  48
#define Q_SCALE 0.14433756729740643f   // 1/sqrt(48)

// ---------------------------------------------------------------------------
// LayerNorm of a (1024 rows x 768), write bf16
// ---------------------------------------------------------------------------
__global__ void k_ln_a(const float* __restrict__ a,
                       const float* __restrict__ w,
                       const float* __restrict__ b,
                       __bf16* __restrict__ an) {
    __shared__ float s1[256], s2[256];
    const int row = blockIdx.x, tid = threadIdx.x;
    const float* x = a + (size_t)row * C_A;
    float x0 = x[tid], x1 = x[tid + 256], x2 = x[tid + 512];
    s1[tid] = x0 + x1 + x2;
    s2[tid] = x0 * x0 + x1 * x1 + x2 * x2;
    __syncthreads();
    for (int o = 128; o > 0; o >>= 1) {
        if (tid < o) { s1[tid] += s1[tid + o]; s2[tid] += s2[tid + o]; }
        __syncthreads();
    }
    const float mean = s1[0] * (1.0f / C_A);
    const float var  = s2[0] * (1.0f / C_A) - mean * mean;
    const float rstd = rsqrtf(var + 1e-5f);
    __bf16* o = an + (size_t)row * C_A;
    o[tid]       = (__bf16)((x0 - mean) * rstd * w[tid]       + b[tid]);
    o[tid + 256] = (__bf16)((x1 - mean) * rstd * w[tid + 256] + b[tid + 256]);
    o[tid + 512] = (__bf16)((x2 - mean) * rstd * w[tid + 512] + b[tid + 512]);
}

// ---------------------------------------------------------------------------
// Transpose + f32->bf16 convert a 768x768 weight: dst[n][k] = src[k][n]
// (makes WMMA B-fragment loads contiguous 32B per lane)
// ---------------------------------------------------------------------------
__global__ void k_wT(const float* __restrict__ src, __bf16* __restrict__ dst) {
    const int idx = blockIdx.x * 256 + threadIdx.x;      // 768*768 elems
    const int kk = idx / C_A, n = idx % C_A;
    dst[(size_t)n * C_A + kk] = (__bf16)src[idx];
}

// ---------------------------------------------------------------------------
// Projections: an[1024,768] @ W[768,768] for Q,K,V,G (mode = blockIdx.z)
// One wave per 16x16 output tile; 24 WMMAs per tile (K=768).
// Q: *1/sqrt(c)+bq -> q[h][i][c] bf16 ; K -> k[h][j][c] bf16
// V: transposed   -> vt[h][c][j] bf16 ; G: sigmoid -> g[i][768] f32
// ---------------------------------------------------------------------------
__global__ void k_proj(const __bf16* __restrict__ an,
                       const __bf16* __restrict__ WqT, const __bf16* __restrict__ WkT,
                       const __bf16* __restrict__ WvT, const __bf16* __restrict__ WgT,
                       const float*  __restrict__ bq,
                       __bf16* __restrict__ qh, __bf16* __restrict__ kh,
                       __bf16* __restrict__ vt, float* __restrict__ g) {
    const int lane = threadIdx.x, hf = lane >> 4, l15 = lane & 15;
    const int n0 = blockIdx.x * 16, m0 = blockIdx.y * 16, mode = blockIdx.z;
    const __bf16* Wt = (mode == 0) ? WqT : (mode == 1) ? WkT : (mode == 2) ? WvT : WgT;

    f32x8 acc = {};
    for (int k0 = 0; k0 < C_A; k0 += 32) {
        bf16x16 amat, bmat;
        // A (16x32 bf16): M = l15, K = hf*16 + e
        const __bf16* ap = an + (size_t)(m0 + l15) * C_A + k0 + hf * 16;
#pragma unroll
        for (int e = 0; e < 8; ++e) { amat[e] = ap[e]; amat[8 + e] = ap[8 + e]; }
        // B (32x16 bf16): column = l15, K = hf*16 + e  (Wt is [n][k])
        const __bf16* bp = Wt + (size_t)(n0 + l15) * C_A + k0 + hf * 16;
#pragma unroll
        for (int e = 0; e < 16; ++e) bmat[e] = bp[e];
        acc = wmma_bf16(amat, bmat, acc);
    }
    // C layout: elem r -> row m0 + hf*8 + r, col n0 + l15
#pragma unroll
    for (int r = 0; r < 8; ++r) {
        const int m = m0 + hf * 8 + r;
        const int n = n0 + l15;
        const int h = n / C_HEAD, cc = n % C_HEAD;
        const float v = acc[r];
        if (mode == 0)      qh[((size_t)h * N_TOK + m) * C_HEAD + cc] = (__bf16)(v * Q_SCALE + bq[n]);
        else if (mode == 1) kh[((size_t)h * N_TOK + m) * C_HEAD + cc] = (__bf16)v;
        else if (mode == 2) vt[((size_t)h * C_HEAD + cc) * N_TOK + m] = (__bf16)v;
        else                g[(size_t)m * C_A + n] = 1.0f / (1.0f + expf(-v));
    }
}

// ---------------------------------------------------------------------------
// bpair = LN(z) @ Wb : pure bandwidth (512MB of z). One wave per (i,j) row.
// Lane holds 4 of 128 elems; butterfly reductions for LN and the 16 outputs.
// ---------------------------------------------------------------------------
__global__ void k_bpair(const float* __restrict__ z,
                        const float* __restrict__ lw, const float* __restrict__ lb,
                        const float* __restrict__ Wb, float* __restrict__ bp) {
    const size_t r = (size_t)((blockIdx.x * blockDim.x + threadIdx.x) >> 5);
    const int lane = threadIdx.x & 31;
    const float* zr = z + r * C_Z;
    float x[4];
#pragma unroll
    for (int e = 0; e < 4; ++e) x[e] = zr[lane * 4 + e];
    float s = x[0] + x[1] + x[2] + x[3];
    float q = x[0] * x[0] + x[1] * x[1] + x[2] * x[2] + x[3] * x[3];
#pragma unroll
    for (int msk = 16; msk > 0; msk >>= 1) { s += __shfl_xor(s, msk); q += __shfl_xor(q, msk); }
    const float mean = s * (1.0f / C_Z);
    const float rstd = rsqrtf(q * (1.0f / C_Z) - mean * mean + 1e-5f);
    float p[NHEAD];
#pragma unroll
    for (int h = 0; h < NHEAD; ++h) p[h] = 0.0f;
#pragma unroll
    for (int e = 0; e < 4; ++e) {
        const int c = lane * 4 + e;
        const float xn = (x[e] - mean) * rstd * lw[c] + lb[c];
        const float* wr = Wb + c * NHEAD;
#pragma unroll
        for (int h = 0; h < NHEAD; ++h) p[h] += xn * wr[h];
    }
#pragma unroll
    for (int h = 0; h < NHEAD; ++h) {
        float t = p[h];
#pragma unroll
        for (int msk = 16; msk > 0; msk >>= 1) t += __shfl_xor(t, msk);
        p[h] = t;
    }
    if (lane == 0) {
        float* o = bp + r * NHEAD;
#pragma unroll
        for (int h = 0; h < NHEAD; ++h) o[h] = p[h];
    }
}

// ---------------------------------------------------------------------------
// Flash attention per (head, 16-query tile). j streamed in blocks of 32.
// scores: 4 WMMAs/block (K=48 padded to 64); PV: 3 WMMAs/block (c=48 = 3x16).
// Per-row (m,l) kept per 16-lane half matching C-layout rows.
// ---------------------------------------------------------------------------
__global__ void k_attn(const __bf16* __restrict__ qh, const __bf16* __restrict__ kh,
                       const __bf16* __restrict__ vt, const float* __restrict__ bp,
                       const float* __restrict__ mask, const float* __restrict__ g,
                       __bf16* __restrict__ og) {
    __shared__ __bf16 pls[16 * 32];   // P tile staging: C-layout f32 -> A-layout bf16
    const int lane = threadIdx.x, hf = lane >> 4, l15 = lane & 15;
    const int i0 = blockIdx.x * 16, h = blockIdx.y;
    const int kbo = hf * 16;

    // Q as two A-fragments (K = 0..31 and 32..63 with 48..63 zero-padded)
    bf16x16 qa0, qa1;
    {
        const __bf16* qp = qh + ((size_t)h * N_TOK + i0 + l15) * C_HEAD;
#pragma unroll
        for (int e = 0; e < 8; ++e) {
            qa0[e]     = qp[kbo + e];
            qa0[8 + e] = qp[kbo + 8 + e];
            qa1[e]     = hf ? (__bf16)0.0f : qp[32 + e];      // 32..47 in half 0
            qa1[8 + e] = hf ? (__bf16)0.0f : qp[40 + e];      // 48..63 padding in half 1
        }
    }

    float m[8], l[8];
    f32x8 o0 = {}, o1 = {}, o2 = {};
#pragma unroll
    for (int r = 0; r < 8; ++r) { m[r] = -1e30f; l[r] = 0.0f; }

    for (int jb = 0; jb < N_TOK; jb += 32) {
        f32x8 st[2];
#pragma unroll
        for (int t = 0; t < 2; ++t) {
            const int j0 = jb + t * 16;
            bf16x16 b0, b1;
            const __bf16* kp = kh + ((size_t)h * N_TOK + j0 + l15) * C_HEAD;
#pragma unroll
            for (int e = 0; e < 16; ++e) {
                b0[e] = kp[hf * 16 + e];                        // K = hf*16+e (<32)
                b1[e] = hf ? (__bf16)0.0f : kp[32 + e];         // K = 32..47, pad 48..63
            }
            f32x8 s = {};
            s = wmma_bf16(qa1, b1, s);
            s = wmma_bf16(qa0, b0, s);
            const int j = j0 + l15;
            const float mval = (-1e9f) * (1.0f - mask[j]);
#pragma unroll
            for (int r = 0; r < 8; ++r) {
                const int i = i0 + hf * 8 + r;
                s[r] += bp[((size_t)i * N_TOK + j) * NHEAD + h] + mval;
            }
            st[t] = s;
        }
        // online softmax (row reductions stay within each 16-lane half)
        float alpha[8];
#pragma unroll
        for (int r = 0; r < 8; ++r) {
            float rm = fmaxf(st[0][r], st[1][r]);
#pragma unroll
            for (int msk = 1; msk < 16; msk <<= 1) rm = fmaxf(rm, __shfl_xor(rm, msk));
            const float nm = fmaxf(m[r], rm);
            const float p0 = expf(st[0][r] - nm);
            const float p1 = expf(st[1][r] - nm);
            float rs = p0 + p1;
#pragma unroll
            for (int msk = 1; msk < 16; msk <<= 1) rs += __shfl_xor(rs, msk);
            const float al = expf(m[r] - nm);
            l[r] = l[r] * al + rs;
            m[r] = nm;
            alpha[r] = al;
            st[0][r] = p0;
            st[1][r] = p1;
        }
#pragma unroll
        for (int r = 0; r < 8; ++r) { o0[r] *= alpha[r]; o1[r] *= alpha[r]; o2[r] *= alpha[r]; }

        // C-layout P -> LDS -> A-layout bf16 fragment
#pragma unroll
        for (int r = 0; r < 8; ++r) {
            pls[(hf * 8 + r) * 32 + l15]      = (__bf16)st[0][r];
            pls[(hf * 8 + r) * 32 + 16 + l15] = (__bf16)st[1][r];
        }
        __syncthreads();
        bf16x16 pa;
        {
            const __bf16* pr = &pls[l15 * 32];
#pragma unroll
            for (int e = 0; e < 8; ++e) { pa[e] = pr[kbo + e]; pa[8 + e] = pr[kbo + 8 + e]; }
        }
        __syncthreads();

        // o[16 x 48] += P(16x32) @ V(32x48): vt is [h][c][j] -> contiguous B loads
#pragma unroll
        for (int t = 0; t < 3; ++t) {
            bf16x16 vb;
            const __bf16* vp = vt + ((size_t)h * C_HEAD + t * 16 + l15) * N_TOK + jb + hf * 16;
#pragma unroll
            for (int e = 0; e < 16; ++e) vb[e] = vp[e];
            if (t == 0)      o0 = wmma_bf16(pa, vb, o0);
            else if (t == 1) o1 = wmma_bf16(pa, vb, o1);
            else             o2 = wmma_bf16(pa, vb, o2);
        }
    }

    // epilogue: o /= l, multiply by gate, store bf16 for final GEMM
#pragma unroll
    for (int r = 0; r < 8; ++r) {
        const int i = i0 + hf * 8 + r;
        const float inv = 1.0f / l[r];
        const size_t base = (size_t)i * C_A + h * C_HEAD;
        og[base + l15]      = (__bf16)(o0[r] * inv * g[base + l15]);
        og[base + 16 + l15] = (__bf16)(o1[r] * inv * g[base + 16 + l15]);
        og[base + 32 + l15] = (__bf16)(o2[r] * inv * g[base + 32 + l15]);
    }
}

// ---------------------------------------------------------------------------
// Final GEMM: (g*o)[1024,768] @ Wout -> f32 output
// ---------------------------------------------------------------------------
__global__ void k_out(const __bf16* __restrict__ og, const __bf16* __restrict__ WoT,
                      float* __restrict__ out) {
    const int lane = threadIdx.x, hf = lane >> 4, l15 = lane & 15;
    const int n0 = blockIdx.x * 16, m0 = blockIdx.y * 16;
    f32x8 acc = {};
    for (int k0 = 0; k0 < C_A; k0 += 32) {
        bf16x16 amat, bmat;
        const __bf16* ap = og + (size_t)(m0 + l15) * C_A + k0 + hf * 16;
#pragma unroll
        for (int e = 0; e < 8; ++e) { amat[e] = ap[e]; amat[8 + e] = ap[8 + e]; }
        const __bf16* bpp = WoT + (size_t)(n0 + l15) * C_A + k0 + hf * 16;
#pragma unroll
        for (int e = 0; e < 16; ++e) bmat[e] = bpp[e];
        acc = wmma_bf16(amat, bmat, acc);
    }
#pragma unroll
    for (int r = 0; r < 8; ++r)
        out[(size_t)(m0 + hf * 8 + r) * C_A + n0 + l15] = acc[r];
}

// ---------------------------------------------------------------------------
extern "C" void kernel_launch(void* const* d_in, const int* in_sizes, int n_in,
                              void* d_out, int out_size, void* d_ws, size_t ws_size,
                              hipStream_t stream) {
    const float* a      = (const float*)d_in[0];
    const float* z      = (const float*)d_in[1];
    const float* mask   = (const float*)d_in[2];
    const float* ln_a_w = (const float*)d_in[3];
    const float* ln_a_b = (const float*)d_in[4];
    const float* ln_z_w = (const float*)d_in[5];
    const float* ln_z_b = (const float*)d_in[6];
    const float* Wq     = (const float*)d_in[7];
    const float* bq     = (const float*)d_in[8];
    const float* Wk     = (const float*)d_in[9];
    const float* Wv     = (const float*)d_in[10];
    const float* Wb     = (const float*)d_in[11];
    const float* Wg     = (const float*)d_in[12];
    const float* Wout   = (const float*)d_in[13];
    float* out = (float*)d_out;

    // workspace carving
    char* ws = (char*)d_ws;
    size_t cur = 0;
    auto carve = [&](size_t bytes) { void* p = ws + cur; cur = (cur + bytes + 255) & ~(size_t)255; return p; };
    __bf16* an   = (__bf16*)carve((size_t)N_TOK * C_A * 2);
    __bf16* WqT  = (__bf16*)carve((size_t)C_A * C_A * 2);
    __bf16* WkT  = (__bf16*)carve((size_t)C_A * C_A * 2);
    __bf16* WvT  = (__bf16*)carve((size_t)C_A * C_A * 2);
    __bf16* WgT  = (__bf16*)carve((size_t)C_A * C_A * 2);
    __bf16* WoT  = (__bf16*)carve((size_t)C_A * C_A * 2);
    __bf16* qh   = (__bf16*)carve((size_t)NHEAD * N_TOK * C_HEAD * 2);
    __bf16* kh   = (__bf16*)carve((size_t)NHEAD * N_TOK * C_HEAD * 2);
    __bf16* vt   = (__bf16*)carve((size_t)NHEAD * C_HEAD * N_TOK * 2);
    float*  g    = (float*) carve((size_t)N_TOK * C_A * 4);
    __bf16* og   = (__bf16*)carve((size_t)N_TOK * C_A * 2);
    float*  bpair= (float*) carve((size_t)N_TOK * N_TOK * NHEAD * 4);
    (void)ws_size; (void)in_sizes; (void)n_in; (void)out_size;

    // 1. LayerNorm(a) -> bf16
    k_ln_a<<<N_TOK, 256, 0, stream>>>(a, ln_a_w, ln_a_b, an);

    // 2. weight transpose+convert
    const int wblocks = (C_A * C_A) / 256;
    k_wT<<<wblocks, 256, 0, stream>>>(Wq, WqT);
    k_wT<<<wblocks, 256, 0, stream>>>(Wk, WkT);
    k_wT<<<wblocks, 256, 0, stream>>>(Wv, WvT);
    k_wT<<<wblocks, 256, 0, stream>>>(Wg, WgT);
    k_wT<<<wblocks, 256, 0, stream>>>(Wout, WoT);

    // 3. q,k,v,g projections (WMMA)
    k_proj<<<dim3(C_A / 16, N_TOK / 16, 4), 32, 0, stream>>>(
        an, WqT, WkT, WvT, WgT, bq, qh, kh, vt, g);

    // 4. bpair = LN(z) @ Wb  (bandwidth-bound: 512MB stream of z)
    k_bpair<<<(N_TOK * N_TOK) / 8, 256, 0, stream>>>(z, ln_z_w, ln_z_b, Wb, bpair);

    // 5. flash attention with pair bias (WMMA)
    k_attn<<<dim3(N_TOK / 16, NHEAD), 32, 0, stream>>>(qh, kh, vt, bpair, mask, g, og);

    // 6. output projection (WMMA) -> f32
    k_out<<<dim3(C_A / 16, N_TOK / 16), 32, 0, stream>>>(og, WoT, out);
}
